// MinGRU_36361193128568
// MI455X (gfx1250) — matrix operand order repaired and checked by reading, not compiled, not run on hardware
//
#include <hip/hip_runtime.h>
#include <hip/hip_bf16.h>

// ---------------------------------------------------------------------------
// MinGRU forward for MI455X (gfx1250, wave32).
//   pre-convert x/W_hg/W_out -> bf16 (memory-bound, keeps GEMM loops pure)
//   GEMM1 (bf16 WMMA, fused softplus epilogue) -> 3-phase chunked log-scan
//   -> GEMM2 (bf16 WMMA) ; h_n emitted by scan phase 2.
// ---------------------------------------------------------------------------

#define B_   4
#define T_   4096
#define DIN  1024
#define H_   1024
#define M1   (B_ * T_)      // 16384 rows for both GEMMs
#define CCH  32             // scan chunks per sequence
#define LCH  (T_ / CCH)     // 128 steps per chunk

typedef __attribute__((ext_vector_type(16))) __bf16 v16bf;
typedef __attribute__((ext_vector_type(8)))  float  v8f;

union FragU {
    v16bf        v;
    unsigned int u[8];
};

// ---- math helpers ---------------------------------------------------------

__device__ __forceinline__ float softplus_f(float x) {
    // stable: max(x,0) + log1p(exp(-|x|))
    return fmaxf(x, 0.0f) + log1pf(__expf(-fabsf(x)));
}

__device__ __forceinline__ float log_g_f(float x) {
    return (x >= 0.0f) ? __logf(x + 0.5f) : -softplus_f(-x);
}

__device__ __forceinline__ float logaddexp_f(float a, float b) {
    float m = fmaxf(a, b);
    if (m == -__builtin_inff()) return m;          // both -inf
    float d = fminf(a, b) - m;                     // <= 0 (or -inf)
    return m + log1pf(__expf(d));
}

__device__ __forceinline__ unsigned int bf16_bits(float f) {
    unsigned int u = __float_as_uint(f);
    return (u + 0x7fffu + ((u >> 16) & 1u)) >> 16;  // round-to-nearest-even
}

__device__ __forceinline__ unsigned int pk2(float lo, float hi) {
    return bf16_bits(lo) | (bf16_bits(hi) << 16);
}

// ---------------------------------------------------------------------------
// f32 -> bf16 bulk conversion (8 elements/thread). Memory-bound; the RNE
// conversion VALU hides under the load/store stream here instead of stalling
// the WMMA pipe inside the GEMM inner loops.
// ---------------------------------------------------------------------------
__global__ __launch_bounds__(256) void cvt_f32_to_bf16(
        const float* __restrict__ src, unsigned short* __restrict__ dst, int n) {
    const size_t base = ((size_t)blockIdx.x * 256 + threadIdx.x) * 8;
    if (base >= (size_t)n) return;
    const float4 a = *reinterpret_cast<const float4*>(src + base);
    const float4 b = *reinterpret_cast<const float4*>(src + base + 4);
    uint4 o;
    o.x = pk2(a.x, a.y);
    o.y = pk2(a.z, a.w);
    o.z = pk2(b.x, b.y);
    o.w = pk2(b.z, b.w);
    *reinterpret_cast<uint4*>(dst + base) = o;
}

// ---- WMMA fragment loaders (wave32 layouts per CDNA5 ISA 7.12.2) ----------

// A matrix 16x32 bf16 (raw ushort storage), row-major, K-contiguous.
// lane L: row = L&15 ; K chunks = [half*8 .. +7] and [16+half*8 .. +7]
__device__ __forceinline__ void loadA_bf16(FragU& f, const unsigned short* __restrict__ tile, int ld) {
    const int lane = threadIdx.x & 31;
    const unsigned short* p = tile + (size_t)(lane & 15) * ld + ((lane >> 4) << 3);
    const uint4 a = *reinterpret_cast<const uint4*>(p);
    const uint4 b = *reinterpret_cast<const uint4*>(p + 16);
    f.u[0] = a.x; f.u[1] = a.y; f.u[2] = a.z; f.u[3] = a.w;
    f.u[4] = b.x; f.u[5] = b.y; f.u[6] = b.z; f.u[7] = b.w;
}

// B matrix 32x16 bf16 from weight stored (outFeature, K) row-major.
// lane L: col = L&15 ; K chunks = [half*8 .. +7] and [16+half*8 .. +7]
__device__ __forceinline__ void loadB_bf16(FragU& f, const unsigned short* __restrict__ tile, int ld) {
    const int lane = threadIdx.x & 31;
    const unsigned short* p = tile + (size_t)(lane & 15) * ld + ((lane >> 4) << 3);
    const uint4 a = *reinterpret_cast<const uint4*>(p);
    const uint4 b = *reinterpret_cast<const uint4*>(p + 16);
    f.u[0] = a.x; f.u[1] = a.y; f.u[2] = a.z; f.u[3] = a.w;
    f.u[4] = b.x; f.u[5] = b.y; f.u[6] = b.z; f.u[7] = b.w;
}

__device__ __forceinline__ v8f wmma_bf16(const FragU& a, const FragU& b, v8f c) {
    return __builtin_amdgcn_wmma_f32_16x16x32_bf16(false, a.v, false, b.v,
                                                   (short)0, c, false, false);
}

// ---------------------------------------------------------------------------
// Kernel 1: hg = x @ W_hg^T with fused elementwise -> lc, lv
// Block = 256 thr (8 waves, 4Mx2N). Block tile: 128 rows x 64 hidden cols
// (each block also computes the paired gate cols, +H_ in W_hg row space).
// grid = (M1/128, H_/64) = (128, 16)
// Inner loop: 12x global_load_b128 + 8x v_wmma per K-step, no conversion VALU.
// ---------------------------------------------------------------------------
__global__ __launch_bounds__(256) void mingru_gemm1_fused(
        const unsigned short* __restrict__ xbf, const unsigned short* __restrict__ whg,
        float* __restrict__ lc, float* __restrict__ lv) {
    const int tid   = threadIdx.x;
    const int wid   = tid >> 5;
    const int lane  = tid & 31;
    const int waveM = wid >> 1;     // 0..3
    const int waveN = wid & 1;      // 0..1
    const int rowBase = blockIdx.x * 128 + waveM * 32;
    const int colBase = blockIdx.y * 64  + waveN * 32;

    v8f accH[2][2], accG[2][2];
#pragma unroll
    for (int mi = 0; mi < 2; ++mi)
#pragma unroll
        for (int ni = 0; ni < 2; ++ni)
#pragma unroll
            for (int i = 0; i < 8; ++i) { accH[mi][ni][i] = 0.0f; accG[mi][ni][i] = 0.0f; }

    for (int k0 = 0; k0 < DIN; k0 += 32) {
        FragU a[2], bh[2], bg[2];
        loadA_bf16(a[0], xbf + (size_t)rowBase * DIN + k0, DIN);
        loadA_bf16(a[1], xbf + (size_t)(rowBase + 16) * DIN + k0, DIN);
        loadB_bf16(bh[0], whg + (size_t)colBase * DIN + k0, DIN);
        loadB_bf16(bh[1], whg + (size_t)(colBase + 16) * DIN + k0, DIN);
        loadB_bf16(bg[0], whg + (size_t)(H_ + colBase) * DIN + k0, DIN);
        loadB_bf16(bg[1], whg + (size_t)(H_ + colBase + 16) * DIN + k0, DIN);
#pragma unroll
        for (int mi = 0; mi < 2; ++mi)
#pragma unroll
            for (int ni = 0; ni < 2; ++ni) {
                accH[mi][ni] = wmma_bf16(a[mi], bh[ni], accH[mi][ni]);
                accG[mi][ni] = wmma_bf16(a[mi], bg[ni], accG[mi][ni]);
            }
    }

    // epilogue: C layout -> elem (m = r + 8*half, n = lane&15)
    const int nl = lane & 15, half = lane >> 4;
#pragma unroll
    for (int mi = 0; mi < 2; ++mi)
#pragma unroll
        for (int ni = 0; ni < 2; ++ni) {
            const int col = colBase + ni * 16 + nl;
#pragma unroll
            for (int r = 0; r < 8; ++r) {
                const int row = rowBase + mi * 16 + r + half * 8;
                const float hd = accH[mi][ni][r];
                const float g  = accG[mi][ni][r];
                const float lcv = -softplus_f(g);
                const float lvv = -softplus_f(-g) + log_g_f(hd);
                const size_t idx = (size_t)row * H_ + col;
                lc[idx] = lcv;
                lv[idx] = lvv;
            }
        }
}

// ---------------------------------------------------------------------------
// Scan phase 1: per-chunk local scan.  thread <-> (b, c, h)
//   SA = sum(lc over chunk), LB = logB within chunk starting from -inf
// grid = B_*H_*CCH / 256 = 512
// ---------------------------------------------------------------------------
__global__ __launch_bounds__(256) void mingru_scan1(
        const float* __restrict__ lc, const float* __restrict__ lv,
        float* __restrict__ SA, float* __restrict__ LB) {
    const int gid = blockIdx.x * 256 + threadIdx.x;
    const int h = gid % H_;
    const int c = (gid / H_) % CCH;
    const int b = gid / (H_ * CCH);
    const size_t base = ((size_t)b * T_ + (size_t)c * LCH) * H_ + h;
    float sa = 0.0f;
    float lb = -__builtin_inff();
    for (int t = 0; t < LCH; ++t) {
        const size_t i = base + (size_t)t * H_;
        const float a = lc[i];
        const float v = lv[i];
        sa += a;
        lb = logaddexp_f(a + lb, v);
    }
    const size_t o = ((size_t)b * CCH + c) * H_ + h;
    SA[o] = sa;
    LB[o] = lb;
}

// ---------------------------------------------------------------------------
// Scan phase 2: cross-chunk scan (32 steps) per channel; writes chunk carry-in
// and the final-state output h_n.  thread <-> (b, h).  grid = 16 x 256
// ---------------------------------------------------------------------------
__global__ __launch_bounds__(256) void mingru_scan2(
        const float* __restrict__ SA, const float* __restrict__ LB,
        float* __restrict__ CI, float* __restrict__ h_n) {
    const int gid = blockIdx.x * 256 + threadIdx.x;   // 0..B_*H_-1
    const int h = gid % H_;
    const int b = gid / H_;
    float carry = -__builtin_inff();
    for (int c = 0; c < CCH; ++c) {
        const size_t i = ((size_t)b * CCH + c) * H_ + h;
        CI[i] = carry;
        carry = logaddexp_f(SA[i] + carry, LB[i]);
    }
    h_n[(size_t)b * H_ + h] = __expf(carry);          // == h[:, -1]
}

// ---------------------------------------------------------------------------
// Scan phase 3: replay chunks with correct carry-in, emit h = exp(logB) in bf16
// grid = 512 x 256, thread <-> (b, c, h)
// ---------------------------------------------------------------------------
__global__ __launch_bounds__(256) void mingru_scan3(
        const float* __restrict__ lc, const float* __restrict__ lv,
        const float* __restrict__ CI, unsigned short* __restrict__ hbf) {
    const int gid = blockIdx.x * 256 + threadIdx.x;
    const int h = gid % H_;
    const int c = (gid / H_) % CCH;
    const int b = gid / (H_ * CCH);
    const size_t base = ((size_t)b * T_ + (size_t)c * LCH) * H_ + h;
    float carry = CI[((size_t)b * CCH + c) * H_ + h];
    for (int t = 0; t < LCH; ++t) {
        const size_t i = base + (size_t)t * H_;
        carry = logaddexp_f(lc[i] + carry, lv[i]);
        hbf[i] = (unsigned short)bf16_bits(__expf(carry));
    }
}

// ---------------------------------------------------------------------------
// Kernel 5: out = h @ W_out^T  (A and B both bf16).  Block tile 128x128,
// 8 waves (4Mx2N), wave tile 32x64.  grid = (128, 8)
// ---------------------------------------------------------------------------
__global__ __launch_bounds__(256) void mingru_gemm2(
        const unsigned short* __restrict__ hbf, const unsigned short* __restrict__ wout,
        float* __restrict__ out) {
    const int tid   = threadIdx.x;
    const int wid   = tid >> 5;
    const int lane  = tid & 31;
    const int waveM = wid >> 1;
    const int waveN = wid & 1;
    const int rowBase = blockIdx.x * 128 + waveM * 32;
    const int colBase = blockIdx.y * 128 + waveN * 64;

    v8f acc[2][4];
#pragma unroll
    for (int mi = 0; mi < 2; ++mi)
#pragma unroll
        for (int ni = 0; ni < 4; ++ni)
#pragma unroll
            for (int i = 0; i < 8; ++i) acc[mi][ni][i] = 0.0f;

    for (int k0 = 0; k0 < H_; k0 += 32) {
        FragU a[2], b[4];
        loadA_bf16(a[0], hbf + (size_t)rowBase * H_ + k0, H_);
        loadA_bf16(a[1], hbf + (size_t)(rowBase + 16) * H_ + k0, H_);
#pragma unroll
        for (int ni = 0; ni < 4; ++ni)
            loadB_bf16(b[ni], wout + (size_t)(colBase + ni * 16) * H_ + k0, H_);
#pragma unroll
        for (int mi = 0; mi < 2; ++mi)
#pragma unroll
            for (int ni = 0; ni < 4; ++ni)
                acc[mi][ni] = wmma_bf16(a[mi], b[ni], acc[mi][ni]);
    }

    const int nl = lane & 15, half = lane >> 4;
#pragma unroll
    for (int mi = 0; mi < 2; ++mi)
#pragma unroll
        for (int ni = 0; ni < 4; ++ni) {
            const int col = colBase + ni * 16 + nl;
#pragma unroll
            for (int r = 0; r < 8; ++r) {
                const int row = rowBase + mi * 16 + r + half * 8;
                out[(size_t)row * H_ + col] = acc[mi][ni][r];
            }
        }
}

// ---------------------------------------------------------------------------

extern "C" void kernel_launch(void* const* d_in, const int* in_sizes, int n_in,
                              void* d_out, int out_size, void* d_ws, size_t ws_size,
                              hipStream_t stream) {
    const float* x    = (const float*)d_in[0];   // (4, 4096, 1024)
    const float* Whg  = (const float*)d_in[1];   // (2048, 1024)
    const float* Wout = (const float*)d_in[2];   // (1024, 1024)

    float* out = (float*)d_out;                          // (4, 4096, 1024)
    float* h_n = out + (size_t)B_ * T_ * H_;             // (4, 1, 1024)

    // workspace layout
    const size_t NE   = (size_t)B_ * T_ * H_;            // 16M elements
    const size_t NWHG = (size_t)2 * H_ * DIN;            // 2M elements
    const size_t NWO  = (size_t)H_ * H_;                 // 1M elements
    float*          lc   = (float*)d_ws;                 // 64 MB
    float*          lv   = lc + NE;                      // 64 MB
    unsigned short* hbf  = (unsigned short*)(lv + NE);   // 32 MB
    unsigned short* xbf  = hbf + NE;                     // 32 MB
    unsigned short* whgb = xbf + NE;                     //  4 MB
    unsigned short* woutb= whgb + NWHG;                  //  2 MB
    float*          SA   = (float*)(woutb + NWO);        // 0.5 MB
    float*          LB   = SA + (size_t)B_ * CCH * H_;   // 0.5 MB
    float*          CI   = LB + (size_t)B_ * CCH * H_;   // 0.5 MB

    // 0) one-shot bf16 conversion of GEMM operands (memory-bound)
    cvt_f32_to_bf16<<<(int)(NE   / (256 * 8)), 256, 0, stream>>>(x,    xbf,   (int)NE);
    cvt_f32_to_bf16<<<(int)(NWHG / (256 * 8)), 256, 0, stream>>>(Whg,  whgb,  (int)NWHG);
    cvt_f32_to_bf16<<<(int)(NWO  / (256 * 8)), 256, 0, stream>>>(Wout, woutb, (int)NWO);

    // 1) fused GEMM1 + elementwise -> lc, lv
    mingru_gemm1_fused<<<dim3(M1 / 128, H_ / 64), 256, 0, stream>>>(xbf, whgb, lc, lv);

    // 2) chunked log-space scan (3 phases), phase 2 also writes h_n
    mingru_scan1<<<(B_ * H_ * CCH) / 256, 256, 0, stream>>>(lc, lv, SA, LB);
    mingru_scan2<<<(B_ * H_) / 256, 256, 0, stream>>>(SA, LB, CI, h_n);
    mingru_scan3<<<(B_ * H_ * CCH) / 256, 256, 0, stream>>>(lc, lv, CI, hbf);

    // 3) GEMM2 -> out
    mingru_gemm2<<<dim3(M1 / 128, H_ / 128), 256, 0, stream>>>(hbf, woutb, out);
}
